// StandardSelfAttention_41257455846000
// MI455X (gfx1250) — matrix operand order repaired and checked by reading, not compiled, not run on hardware
//
#include <hip/hip_runtime.h>
#include <hip/hip_bf16.h>

// ---------------------------------------------------------------------------
// Causal flash-attention, B=2,H=16,T=2048,D=64, fp32 in/out.
// S^T = K x Q^T, O^T = V^T x P^T (v_wmma_f32_16x16x32_bf16); per-lane softmax
// stats; row sums on the tensor core. Workspace path: one-time bf16 pre-pack
// of K / V^T, then the main loop streams K/V blocks into double-buffered LDS
// with global_load_async_to_lds_b128 (ASYNCcnt), overlapping fill with WMMA.
// ---------------------------------------------------------------------------

typedef __attribute__((ext_vector_type(16))) __bf16 v16bf;
typedef __attribute__((ext_vector_type(8)))  float  v8f;

#define ATT_B 2
#define ATT_H 16
#define ATT_T 2048
#define ATT_D 64
#define ATT_BH (ATT_B * ATT_H)
#define BLOCK_M 128   // query rows per workgroup
#define BLOCK_N 64    // keys per iteration
#define NWAVE   8

union BF16Frag { unsigned u[8]; v16bf v; };

__device__ __forceinline__ unsigned packbf_rz(float lo, float hi) {
    return __builtin_amdgcn_perm(__float_as_uint(hi), __float_as_uint(lo),
                                 0x07060302u);
}
__device__ __forceinline__ unsigned f2bf_rne(float f) {
    unsigned u = __float_as_uint(f);
    u = u + 0x7FFFu + ((u >> 16) & 1u);
    return u >> 16;
}

// A-fragment K-index base for 16-bit 16x32 A matrix (contiguous k = 16h + s, matching B).
__device__ __forceinline__ int a_k0(int i, int half) {
    return (half ? 16 : 0) + 2 * i;
}
// B-fragment K-index base for 16-bit 32x16 B matrix.
__device__ __forceinline__ int b_k0(int i, int half) {
    return (half ? 16 : 0) + 2 * i;
}

// ---------------------------------------------------------------------------
// Pre-pack: K -> bf16 [bh][t][d]; V -> bf16 transposed [bh][d][t].
// ---------------------------------------------------------------------------
__global__ __launch_bounds__(256) void prepack_kv(
    const float* __restrict__ K, const float* __restrict__ V,
    unsigned short* __restrict__ Kbf, unsigned short* __restrict__ Vt)
{
    __shared__ unsigned short sT[ATT_D * 64];

    const int bh    = blockIdx.y;
    const int kbase = blockIdx.x * 64;
    const int tid   = threadIdx.x;

    const float* Kg = K + (size_t)bh * ATT_T * ATT_D;
    const float* Vg = V + (size_t)bh * ATT_T * ATT_D;
    unsigned short* Kb = Kbf + (size_t)bh * ATT_T * ATT_D;
    unsigned short* Vo = Vt  + (size_t)bh * ATT_T * ATT_D;

    for (int e = tid; e < 64 * (ATT_D / 2); e += 256) {
        int kk = e >> 5, dp = (e & 31) << 1;
        float2 kv = *(const float2*)&Kg[(size_t)(kbase + kk) * ATT_D + dp];
        *(unsigned*)&Kb[(size_t)(kbase + kk) * ATT_D + dp] = packbf_rz(kv.x, kv.y);
    }
    for (int e = tid; e < 64 * (ATT_D / 2); e += 256) {
        int kk = e >> 5, dp = (e & 31) << 1;
        float2 vv = *(const float2*)&Vg[(size_t)(kbase + kk) * ATT_D + dp];
        sT[dp * 64 + kk]       = (unsigned short)(__float_as_uint(vv.x) >> 16);
        sT[(dp + 1) * 64 + kk] = (unsigned short)(__float_as_uint(vv.y) >> 16);
    }
    __syncthreads();
    for (int e = tid; e < ATT_D * 32; e += 256) {
        int d = e >> 5, kp = (e & 31) << 1;
        *(unsigned*)&Vo[(size_t)d * ATT_T + kbase + kp] =
            *(const unsigned*)&sT[d * 64 + kp];
    }
}

// ---------------------------------------------------------------------------
// Async LDS fill (workspace path): 4 x global_load_async_to_lds_b128/thread.
// ---------------------------------------------------------------------------
__device__ __forceinline__ void async_fill(
    uint4* dK, uint4* dV,
    const unsigned short* __restrict__ Kb,
    const unsigned short* __restrict__ Vtb,
    int kbase, int tid)
{
    // K block is 8 KB contiguous; INST_OFFSET advances both LDS and global.
    {
        unsigned long long src =
            (unsigned long long)(size_t)((const char*)(Kb + (size_t)kbase * ATT_D)
                                         + tid * 16);
        unsigned lds = (unsigned)(size_t)((char*)dK + tid * 16);
        asm volatile("global_load_async_to_lds_b128 %0, %1, off"
                     :: "v"(lds), "v"(src) : "memory");
        asm volatile("global_load_async_to_lds_b128 %0, %1, off offset:4096"
                     :: "v"(lds), "v"(src) : "memory");
    }
    // V^T: 64 rows x 128 B, global row stride T*2 bytes.
    {
        int e0 = tid,        d0 = e0 >> 3, j0 = e0 & 7;
        int e1 = tid + 256,  d1 = e1 >> 3, j1 = e1 & 7;
        unsigned long long s0 = (unsigned long long)(size_t)
            ((const char*)(Vtb + (size_t)d0 * ATT_T + kbase) + j0 * 16);
        unsigned long long s1 = (unsigned long long)(size_t)
            ((const char*)(Vtb + (size_t)d1 * ATT_T + kbase) + j1 * 16);
        unsigned l0 = (unsigned)(size_t)((char*)dV + e0 * 16);
        unsigned l1 = (unsigned)(size_t)((char*)dV + e1 * 16);
        asm volatile("global_load_async_to_lds_b128 %0, %1, off"
                     :: "v"(l0), "v"(s0) : "memory");
        asm volatile("global_load_async_to_lds_b128 %0, %1, off"
                     :: "v"(l1), "v"(s1) : "memory");
    }
}

// ---------------------------------------------------------------------------
// One 64-key compute step (S^T orientation).
// ---------------------------------------------------------------------------
template <bool MASKED>
__device__ __forceinline__ void compute_step(
    int kbase, const unsigned short* sK, const unsigned short* sV,
    unsigned short* sPw, const BF16Frag* qfrag, const BF16Frag& ones,
    v8f* acc, float& m_run, float& l_run,
    int wave, int half, int ln, int q_tile)
{
    if (MASKED && (kbase > q_tile + wave * 16 + 15)) return;   // fully future

    // ---- S^T tiles: st[nt] = K_tile(16x64) x Q^T(64x16), 8 WMMAs ----
    v8f st[4];
#pragma unroll
    for (int nt = 0; nt < 4; ++nt) {
        v8f s = (v8f){};
#pragma unroll
        for (int c = 0; c < 2; ++c) {
            BF16Frag a;
#pragma unroll
            for (int i = 0; i < 8; ++i) {
                int d0 = c * 32 + a_k0(i, half);
                a.u[i] = *(const unsigned*)&sK[(nt * 16 + ln) * ATT_D + d0];
            }
            s = __builtin_amdgcn_wmma_f32_16x16x32_bf16(
                    false, a.v, false, qfrag[c].v, (short)0, s, false, false);
        }
        st[nt] = s;
    }

    if (MASKED) {
        const int qrel = q_tile + wave * 16 + ln - kbase;
#pragma unroll
        for (int nt = 0; nt < 4; ++nt)
#pragma unroll
            for (int r = 0; r < 8; ++r)
                st[nt][r] = (nt * 16 + r + 8 * half <= qrel) ? st[nt][r]
                                                             : -1e30f;
    }

    // ---- Per-lane row max over 32 values + one cross-half shuffle ----
    float tm[4];
#pragma unroll
    for (int nt = 0; nt < 4; ++nt) {
        float a0 = fmaxf(fmaxf(st[nt][0], st[nt][1]),
                         fmaxf(st[nt][2], st[nt][3]));
        float a1 = fmaxf(fmaxf(st[nt][4], st[nt][5]),
                         fmaxf(st[nt][6], st[nt][7]));
        tm[nt] = fmaxf(a0, a1);
    }
    float mx = fmaxf(fmaxf(tm[0], tm[1]), fmaxf(tm[2], tm[3]));
    mx = fmaxf(mx, __shfl_xor(mx, 16, 32));

    const float mnew = fmaxf(m_run, mx);
    const float corr = exp2f(m_run - mnew);
    m_run = mnew;
    const float lc = l_run * corr;

    // ---- P^T = exp2(S^T - m): pack adjacent-key pairs, b32 LDS stores ----
#pragma unroll
    for (int nt = 0; nt < 4; ++nt)
#pragma unroll
        for (int i = 0; i < 4; ++i) {
            float p0 = exp2f(st[nt][2 * i]     - mnew);
            float p1 = exp2f(st[nt][2 * i + 1] - mnew);
            *(unsigned*)&sPw[ln * BLOCK_N + nt * 16 + 8 * half + 2 * i] =
                packbf_rz(p0, p1);
        }
    asm volatile("s_wait_dscnt 0" ::: "memory");

    BF16Frag pT[2];
#pragma unroll
    for (int c = 0; c < 2; ++c)
#pragma unroll
        for (int i = 0; i < 8; ++i) {
            int k0 = c * 32 + b_k0(i, half);
            pT[c].u[i] = *(const unsigned*)&sPw[ln * BLOCK_N + k0];
        }

    // ---- Row sums on tensor core: ones(16x32) x P^T ----
    v8f sr = (v8f){};
    sr = __builtin_amdgcn_wmma_f32_16x16x32_bf16(
             false, ones.v, false, pT[0].v, (short)0, sr, false, false);
    sr = __builtin_amdgcn_wmma_f32_16x16x32_bf16(
             false, ones.v, false, pT[1].v, (short)0, sr, false, false);
    l_run = lc + sr[0];

    // ---- Rescale accumulator, then O^T += V^T x P^T, 8 WMMAs ----
#pragma unroll
    for (int f = 0; f < 4; ++f) acc[f] *= corr;
#pragma unroll
    for (int f = 0; f < 4; ++f)
#pragma unroll
        for (int c = 0; c < 2; ++c) {
            BF16Frag a;
#pragma unroll
            for (int i = 0; i < 8; ++i) {
                int kk = c * 32 + a_k0(i, half);
                a.u[i] = *(const unsigned*)&sV[(f * 16 + ln) * BLOCK_N + kk];
            }
            acc[f] = __builtin_amdgcn_wmma_f32_16x16x32_bf16(
                         false, a.v, false, pT[c].v, (short)0,
                         acc[f], false, false);
        }
}

// ---------------------------------------------------------------------------
// Main kernel
// ---------------------------------------------------------------------------
template <bool USE_WS>
__global__ __launch_bounds__(256) void fa_causal_bf16_wmma(
    const float* __restrict__ Q,
    const float* __restrict__ K,
    const float* __restrict__ V,
    const unsigned short* __restrict__ Kbf,
    const unsigned short* __restrict__ Vt,
    float* __restrict__ O)
{
    __shared__ uint4 sKbuf[2][512];            // 2 x 8 KB bf16 [key][d]
    __shared__ uint4 sVbuf[2][512];            // 2 x 8 KB bf16 [d][key]
    __shared__ uint4 sPq[NWAVE * 128];         // 16 KB per-wave P^T [m][key]

    const int tid  = threadIdx.x;
    const int wave = tid >> 5;
    const int lane = tid & 31;
    const int half = lane >> 4;
    const int ln   = lane & 15;
    unsigned short* sPw = (unsigned short*)(sPq + wave * 128);

    const int bh     = blockIdx.y;
    const int q_tile = blockIdx.x * BLOCK_M;

    const float* Qg = Q + (size_t)bh * ATT_T * ATT_D;
    const float* Kg = K + (size_t)bh * ATT_T * ATT_D;
    const float* Vg = V + (size_t)bh * ATT_T * ATT_D;
    float*       Og = O + (size_t)bh * ATT_T * ATT_D;
    const unsigned short* Kb  = USE_WS ? Kbf + (size_t)bh * ATT_T * ATT_D : nullptr;
    const unsigned short* Vtb = USE_WS ? Vt  + (size_t)bh * ATT_T * ATT_D : nullptr;

    // ---- Q^T B-fragments; fold 1/8 * log2(e) into Q ----
    const float qscale = 0.125f * 1.4426950408889634f;
    BF16Frag qfrag[2];
    {
        const float* qp = Qg + (size_t)(q_tile + wave * 16 + ln) * ATT_D;
#pragma unroll
        for (int c = 0; c < 2; ++c)
#pragma unroll
            for (int i = 0; i < 8; ++i) {
                int d0 = c * 32 + b_k0(i, half);
                qfrag[c].u[i] = (f2bf_rne(qp[d0 + 1] * qscale) << 16) |
                                 f2bf_rne(qp[d0] * qscale);
            }
    }

    BF16Frag ones;
#pragma unroll
    for (int i = 0; i < 8; ++i) ones.u[i] = 0x3F803F80u;

    v8f acc[4];
#pragma unroll
    for (int f = 0; f < 4; ++f) acc[f] = (v8f){};
    float m_run = -1e30f, l_run = 0.0f;

    const int n_full = q_tile >> 6;
    const int ntot   = n_full + 2;

    if (USE_WS) {
        // ---- Async double-buffered pipeline ----
        async_fill(sKbuf[0], sVbuf[0], Kb, Vtb, 0, tid);
        for (int kb = 0; kb < ntot; ++kb) {
            asm volatile("s_wait_asynccnt 0x0" ::: "memory");
            __syncthreads();
            if (kb + 1 < ntot)
                async_fill(sKbuf[(kb + 1) & 1], sVbuf[(kb + 1) & 1],
                           Kb, Vtb, (kb + 1) * BLOCK_N, tid);
            const unsigned short* cK = (const unsigned short*)sKbuf[kb & 1];
            const unsigned short* cV = (const unsigned short*)sVbuf[kb & 1];
            if (kb < n_full)
                compute_step<false>(kb * BLOCK_N, cK, cV, sPw, qfrag, ones,
                                    acc, m_run, l_run, wave, half, ln, q_tile);
            else
                compute_step<true>(kb * BLOCK_N, cK, cV, sPw, qfrag, ones,
                                   acc, m_run, l_run, wave, half, ln, q_tile);
        }
    } else {
        // ---- Fallback: synchronous convert-in-loop, single buffer ----
        unsigned short* sK = (unsigned short*)sKbuf[0];
        unsigned short* sV = (unsigned short*)sVbuf[0];
        for (int kb = 0; kb < ntot; ++kb) {
            const int kbase = kb * BLOCK_N;
            for (int e = tid; e < BLOCK_N * (ATT_D / 2); e += 256) {
                int kk = e >> 5, dp = (e & 31) << 1;
                float2 kv = *(const float2*)&Kg[(size_t)(kbase + kk) * ATT_D + dp];
                *(unsigned*)&sK[kk * ATT_D + dp] = packbf_rz(kv.x, kv.y);
            }
            for (int e = tid; e < (BLOCK_N / 2) * (ATT_D / 2); e += 256) {
                int kk = (e & 31) << 1, dp = (e >> 5) << 1;
                float2 v0 = *(const float2*)&Vg[(size_t)(kbase + kk) * ATT_D + dp];
                float2 v1 = *(const float2*)&Vg[(size_t)(kbase + kk + 1) * ATT_D + dp];
                *(unsigned*)&sV[dp * BLOCK_N + kk]       = packbf_rz(v0.x, v1.x);
                *(unsigned*)&sV[(dp + 1) * BLOCK_N + kk] = packbf_rz(v0.y, v1.y);
            }
            __syncthreads();
            if (kb < n_full)
                compute_step<false>(kbase, sK, sV, sPw, qfrag, ones,
                                    acc, m_run, l_run, wave, half, ln, q_tile);
            else
                compute_step<true>(kbase, sK, sV, sPw, qfrag, ones,
                                   acc, m_run, l_run, wave, half, ln, q_tile);
            __syncthreads();
        }
    }

    // ---- Epilogue: O^T lane columns -> contiguous float4 row stores ----
    const float inv = 1.0f / l_run;
    const int row = q_tile + wave * 16 + ln;
#pragma unroll
    for (int f = 0; f < 4; ++f) {
        float4 o0, o1;
        o0.x = acc[f][0] * inv; o0.y = acc[f][1] * inv;
        o0.z = acc[f][2] * inv; o0.w = acc[f][3] * inv;
        o1.x = acc[f][4] * inv; o1.y = acc[f][5] * inv;
        o1.z = acc[f][6] * inv; o1.w = acc[f][7] * inv;
        const int d0 = f * 16 + 8 * half;
        *(float4*)&Og[(size_t)row * ATT_D + d0]     = o0;
        *(float4*)&Og[(size_t)row * ATT_D + d0 + 4] = o1;
    }
}

extern "C" void kernel_launch(void* const* d_in, const int* in_sizes, int n_in,
                              void* d_out, int out_size, void* d_ws, size_t ws_size,
                              hipStream_t stream) {
    const float* Q = (const float*)d_in[0];
    const float* K = (const float*)d_in[1];
    const float* V = (const float*)d_in[2];
    float* O = (float*)d_out;

    dim3 grid(ATT_T / BLOCK_M, ATT_BH);   // (16, 32)
    dim3 block(256);

    const size_t elems = (size_t)ATT_BH * ATT_T * ATT_D;
    const size_t need  = 2 * elems * sizeof(unsigned short);   // 16 MiB

    if (d_ws && ws_size >= need) {
        unsigned short* Kbf = (unsigned short*)d_ws;
        unsigned short* Vt  = Kbf + elems;
        prepack_kv<<<dim3(ATT_T / 64, ATT_BH), block, 0, stream>>>(K, V, Kbf, Vt);
        fa_causal_bf16_wmma<true><<<grid, block, 0, stream>>>(Q, K, V, Kbf, Vt, O);
    } else {
        fa_causal_bf16_wmma<false><<<grid, block, 0, stream>>>(Q, K, V,
                                                               nullptr, nullptr, O);
    }
}
